// DIN_1039382086155
// MI455X (gfx1250) — matrix operand order repaired and checked by reading, not compiled, not run on hardware
//
#include <hip/hip_runtime.h>

// DIN attention: query[B,D], facts[B,T,D], mask[B,T](u8), W1[4D,H], b1[H], W2[H,1], b2[1]
// out[B,D].  B=4096, T=200, D=128, H=16.
#define B_  4096
#define T_  200
#define D_  128
#define H_  16
#define NEG_ (-4294967295.0f)

typedef __attribute__((ext_vector_type(16))) __bf16 v16bf;
typedef __attribute__((ext_vector_type(8)))  float  v8f;

constexpr int TPAD  = 132;              // facts row stride in LDS (pad 128->132 floats, bank-conflict free)
constexpr int NTILE = 13;               // ceil(200/16)

// LDS byte offsets
constexpr int OFF_FACTS = 0;                       // 200*132*4 = 105600
constexpr int OFF_Q     = 105600;                  // 128*4     = 512
constexpr int OFF_WM    = 106112;                  // 256*16*2  = 8192 (bf16 merged weights)
constexpr int OFF_QB    = 114304;                  // 16*4
constexpr int OFF_W2    = 114368;                  // 17*4 -> pad 80  (w2[16] + b2)
constexpr int OFF_SC    = 114448;                  // 208*4 = 832 (scores / attn)
constexpr int OFF_RED   = 115280;                  // 8*4 = 32
constexpr int OFF_OUT   = 115312;                  // 128*4 = 512
constexpr int SMEM_BYTES = 115824;

__global__ __launch_bounds__(256, 1)
void din_attn_kernel(const float* __restrict__ query,
                     const float* __restrict__ facts,
                     const unsigned char* __restrict__ mask,
                     const float* __restrict__ W1,
                     const float* __restrict__ b1,
                     const float* __restrict__ W2,
                     const float* __restrict__ b2,
                     float* __restrict__ out)
{
    extern __shared__ char smem[];
    float*  facts_s = (float*)(smem + OFF_FACTS);
    float*  q_s     = (float*)(smem + OFF_Q);
    __bf16* wm_s    = (__bf16*)(smem + OFF_WM);
    float*  qb_s    = (float*)(smem + OFF_QB);
    float*  w2_s    = (float*)(smem + OFF_W2);
    float*  sc_s    = (float*)(smem + OFF_SC);
    float*  red_s   = (float*)(smem + OFF_RED);
    float*  oa_s    = (float*)(smem + OFF_OUT);

    const int tid  = threadIdx.x;
    const int lane = tid & 31;
    const int wid  = tid >> 5;
    const int b    = blockIdx.x;

    const float* gq = query + (size_t)b * D_;
    const float* gf = facts + (size_t)b * (size_t)T_ * D_;

    // ---- Phase 0: async-stage facts[b] (f32, padded rows) into LDS -------------
    // 200 rows * 32 float4 = 6400 b128 transfers, 25 per thread, tracked by ASYNCcnt.
    for (int i = tid; i < T_ * (D_ / 4); i += 256) {
        int t = i >> 5, seg = i & 31;
        const float* g = gf + (size_t)t * D_ + seg * 4;
        unsigned lds_off = (unsigned)(size_t)(facts_s + t * TPAD + seg * 4);
        asm volatile("global_load_async_to_lds_b128 %0, %1, off"
                     :: "v"(lds_off), "v"(g) : "memory");
    }

    // ---- Phase 1: small tables into LDS ---------------------------------------
    if (tid < D_) q_s[tid] = gq[tid];
    if (tid < H_) { qb_s[tid] = b1[tid]; w2_s[tid] = W2[tid]; }
    if (tid == H_) w2_s[H_] = b2[0];
    // Merged weights Wm[256][16] (bf16):
    //   k<128 : W1b - W1c  (facts part minus (q-f) part)
    //   k>=128: W1d        ((q*f) part)
    for (int i = tid; i < 256 * H_; i += 256) {
        int k = i >> 4, h = i & 15;
        float wv = (k < 128)
                 ? (W1[(128 + k) * H_ + h] - W1[(256 + k) * H_ + h])
                 :  W1[(384 + (k - 128)) * H_ + h];
        wm_s[i] = (__bf16)wv;
    }
    __syncthreads();

    // ---- Phase 2: per-row hidden bias  qb[h] = b1[h] + sum_k q[k]*(W1a+W1c)[k][h]
    {
        int h = tid & 15, k0 = (tid >> 4) * 8;
        float p = 0.0f;
        #pragma unroll
        for (int j = 0; j < 8; ++j) {
            int k = k0 + j;
            p += q_s[k] * (W1[k * H_ + h] + W1[(256 + k) * H_ + h]);
        }
        atomicAdd(&qb_s[h], p);
    }

    // wait for async facts + LDS atomics, then full barrier
    asm volatile("s_wait_asynccnt 0" ::: "memory");
    __syncthreads();

    const float b2v = w2_s[H_];

    // ---- Phase 3: score GEMM via v_wmma_f32_16x16x32_bf16 ---------------------
    // Preload B fragments (Wm, K=256 -> 8 fragments of K=32), identical for all tiles.
    // B 32x16 bf16 k order used here (contiguous, matches A): lane n=lane%16 holds column n; lanes 0-15: K=0..15, 16-31: K=16..31.
    const int n    = lane & 15;
    const int kb16 = (lane >> 4) * 16;
    v16bf bfr[8];
    #pragma unroll
    for (int kk = 0; kk < 8; ++kk) {
        #pragma unroll
        for (int e = 0; e < 16; ++e)
            bfr[kk][e] = wm_s[(kk * 32 + kb16 + e) * H_ + n];
    }

    const int kbase = (lane >> 4) * 16;  // A k order used here (contiguous, matches B): lanes 0-15 K 0-15, lanes 16-31 K 16-31
    for (int tile = wid; tile < NTILE; tile += 8) {
        const int m = lane & 15;
        int t = tile * 16 + m; if (t > T_ - 1) t = T_ - 1;     // clamp pad rows (masked later)
        const float* frow = facts_s + t * TPAD;

        v8f acc = {};
        #pragma unroll
        for (int kk = 0; kk < 8; ++kk) {
            v16bf a;
            #pragma unroll
            for (int e = 0; e < 16; ++e) {
                int Kl = kbase + e;
                int Kg = kk * 32 + Kl;
                float vv = (Kg < 128) ? frow[Kg]
                                      : frow[Kg - 128] * q_s[Kg - 128];
                a[e] = (__bf16)vv;
            }
            acc = __builtin_amdgcn_wmma_f32_16x16x32_bf16(
                      false, a, false, bfr[kk], (short)0, acc, false, false);
        }

        // acc layout: lane holds h=n, VGPR v holds row m = v + 8*(lane/16).
        // score[t] = sum_h sigmoid(acc + qb[h]) * W2[h] + b2  -> 16-lane shuffle reduce.
        #pragma unroll
        for (int v = 0; v < 8; ++v) {
            float x = acc[v] + qb_s[n];
            float hs = 1.0f / (1.0f + __expf(-x));
            float c = hs * w2_s[n];
            c += __shfl_xor(c, 1, 32);
            c += __shfl_xor(c, 2, 32);
            c += __shfl_xor(c, 4, 32);
            c += __shfl_xor(c, 8, 32);
            if (n == 0) {
                int mrow = v + ((lane >> 4) << 3);
                sc_s[tile * 16 + mrow] = c + b2v;
            }
        }
    }
    __syncthreads();

    // ---- Phase 4: masked softmax over T ---------------------------------------
    float s = NEG_;
    if (tid < T_) s = mask[(size_t)b * T_ + tid] ? sc_s[tid] : NEG_;
    float mx = s;
    mx = fmaxf(mx, __shfl_xor(mx, 16, 32));
    mx = fmaxf(mx, __shfl_xor(mx,  8, 32));
    mx = fmaxf(mx, __shfl_xor(mx,  4, 32));
    mx = fmaxf(mx, __shfl_xor(mx,  2, 32));
    mx = fmaxf(mx, __shfl_xor(mx,  1, 32));
    if (lane == 0) red_s[wid] = mx;
    __syncthreads();
    if (tid == 0) {
        float mm = red_s[0];
        #pragma unroll
        for (int i = 1; i < 8; ++i) mm = fmaxf(mm, red_s[i]);
        red_s[0] = mm;
    }
    __syncthreads();
    const float bmax = red_s[0];
    __syncthreads();

    float ev = (tid < T_) ? __expf(s - bmax) : 0.0f;
    float se = ev;
    se += __shfl_xor(se, 16, 32);
    se += __shfl_xor(se,  8, 32);
    se += __shfl_xor(se,  4, 32);
    se += __shfl_xor(se,  2, 32);
    se += __shfl_xor(se,  1, 32);
    if (lane == 0) red_s[wid] = se;
    __syncthreads();
    if (tid == 0) {
        float ss = 0.0f;
        #pragma unroll
        for (int i = 0; i < 8; ++i) ss += red_s[i];
        red_s[0] = ss;
    }
    __syncthreads();
    const float ssum = red_s[0];
    if (tid < T_) sc_s[tid] = ev / ssum;     // attn weights
    if (tid < D_) oa_s[tid] = 0.0f;
    __syncthreads();

    // ---- Phase 5: out[b,d] = sum_t attn[t] * facts[t,d] (all from LDS) --------
    {
        int d = tid & 127, half = tid >> 7;
        float acc2 = 0.0f;
        for (int t = half * 100; t < half * 100 + 100; ++t)
            acc2 += sc_s[t] * facts_s[t * TPAD + d];
        atomicAdd(&oa_s[d], acc2);
    }
    __syncthreads();
    if (tid < D_) out[(size_t)b * D_ + tid] = oa_s[tid];
}

extern "C" void kernel_launch(void* const* d_in, const int* in_sizes, int n_in,
                              void* d_out, int out_size, void* d_ws, size_t ws_size,
                              hipStream_t stream) {
    const float*         query = (const float*)d_in[0];
    const float*         facts = (const float*)d_in[1];
    const unsigned char* mask  = (const unsigned char*)d_in[2];
    const float*         W1    = (const float*)d_in[3];
    const float*         b1    = (const float*)d_in[4];
    const float*         W2    = (const float*)d_in[5];
    const float*         b2    = (const float*)d_in[6];
    float*               out   = (float*)d_out;

    hipFuncSetAttribute((const void*)din_attn_kernel,
                        hipFuncAttributeMaxDynamicSharedMemorySize, SMEM_BYTES);
    din_attn_kernel<<<B_, 256, SMEM_BYTES, stream>>>(query, facts, mask, W1, b1, W2, b2, out);
}
